// GNNPolicy_1692217115507
// MI455X (gfx1250) — matrix operand order repaired and checked by reading, not compiled, not run on hardware
//
#include <hip/hip_runtime.h>
#include <cstdint>

typedef __attribute__((ext_vector_type(16))) _Float16 v16h;
typedef __attribute__((ext_vector_type(8)))  float    v8f;

// ---------- helpers ----------

__device__ __forceinline__ float waveSum(float v) {
#pragma unroll
  for (int m = 16; m > 0; m >>= 1) v += __shfl_xor(v, m, 32);
  return v;
}

// float atomic max via monotone int/uint encodings (works for mixed signs,
// init value must be -inf). Lowers to global_atomic_max_i32 / min_u32.
__device__ __forceinline__ void atomMaxF(float* addr, float val) {
  if (val >= 0.0f) atomicMax((int*)addr, __float_as_int(val));
  else             atomicMin((unsigned int*)addr, __float_as_uint(val));
}

__device__ __forceinline__ void atomAddF(float* addr, float val) {
  unsafeAtomicAdd(addr, val);   // native global_atomic_add_f32 on gfx1250
}

// ---------- 1) node encoder: Linear(5->64) + ReLU + LayerNorm ----------
// one wave per node; lane l owns columns l and l+32.
__global__ void enc_kernel(const float* __restrict__ x, const float* __restrict__ Wenc,
                           const float* __restrict__ benc, const float* __restrict__ gln,
                           const float* __restrict__ bln, float* __restrict__ h, int n) {
  int wave = threadIdx.x >> 5, lane = threadIdx.x & 31;
  int node = blockIdx.x * 8 + wave;
  if (node >= n) return;
  float xf[5];
#pragma unroll
  for (int f = 0; f < 5; ++f) xf[f] = x[node * 5 + f];
  int c0 = lane, c1 = lane + 32;
  float v0 = benc[c0], v1 = benc[c1];
#pragma unroll
  for (int f = 0; f < 5; ++f) {
    v0 = fmaf(xf[f], Wenc[f * 64 + c0], v0);
    v1 = fmaf(xf[f], Wenc[f * 64 + c1], v1);
  }
  v0 = fmaxf(v0, 0.0f); v1 = fmaxf(v1, 0.0f);
  float s  = waveSum(v0 + v1);
  float s2 = waveSum(v0 * v0 + v1 * v1);
  float mu  = s * (1.0f / 64.0f);
  float var = s2 * (1.0f / 64.0f) - mu * mu;
  float inv = rsqrtf(var + 1e-5f);
  h[node * 64 + c0] = (v0 - mu) * inv * gln[c0] + bln[c0];
  h[node * 64 + c1] = (v1 - mu) * inv * gln[c1] + bln[c1];
}

// ---------- 2) ht = h @ W  ([n,64] x [64,64]) via v_wmma_f32_16x16x32_f16 ----------
// 256 threads = 8 waves; 32 rows/block; wave w -> row tile (w>>2), col tile (w&3).
// A tile loaded as float4 (global_load_b128), output staged through LDS and
// written back as float4 (global_store_b128) -- no per-element EXEC branching.
__global__ void gemm64_wmma(const float* __restrict__ Hin, const float* __restrict__ W,
                            float* __restrict__ Out, int n) {
  __shared__ _Float16 sWt[64 * 64];  // W transposed: sWt[c*64+k] = W[k*64+c]
  __shared__ _Float16 sA[32 * 64];   // sA[r*64+k]
  __shared__ float    sOut[32 * 64]; // output tile staging
  int tid = threadIdx.x;
  int rowBase = blockIdx.x * 32;
  bool full = (rowBase + 32 <= n);

  // stage W (coalesced global read, transposed LDS write)
  for (int i = tid; i < 64 * 64; i += 256) {
    int k = i >> 6, c = i & 63;
    sWt[c * 64 + k] = (_Float16)W[i];
  }
  // stage A rows as f16
  if (full) {
    const float4* H4 = (const float4*)(Hin + (size_t)rowBase * 64);
    for (int i = tid; i < 32 * 16; i += 256) {   // 512 float4, coalesced
      float4 v = H4[i];
      int b = i * 4;
      sA[b + 0] = (_Float16)v.x; sA[b + 1] = (_Float16)v.y;
      sA[b + 2] = (_Float16)v.z; sA[b + 3] = (_Float16)v.w;
    }
  } else {
    for (int i = tid; i < 32 * 64; i += 256) {
      int r = i >> 6, k = i & 63;
      int gr = rowBase + r;
      sA[i] = (_Float16)((gr < n) ? Hin[(size_t)gr * 64 + k] : 0.0f);
    }
  }
  __syncthreads();

  int wave = tid >> 5, lane = tid & 31;
  int rtile = (wave >> 2) * 16;   // 0 or 16
  int ctile = (wave & 3) * 16;    // 0,16,32,48
  int half = lane >> 4;           // 0/1
  int r = lane & 15;

  v8f c = {};
#pragma unroll
  for (int kk = 0; kk < 64; kk += 32) {
    v16h a, b;
    // A fragment (16x32 f16): K = kk + half*16 + i, row = rtile + r.
    int arow = rtile + r;
#pragma unroll
    for (int i = 0; i < 16; ++i) {
      int k = kk + half * 16 + i;
      a[i] = sA[arow * 64 + k];
    }
    // B fragment (32x16 f16): K = kk + half*16 + i, column = ctile + r.
    int bcol = ctile + r;
#pragma unroll
    for (int i = 0; i < 16; ++i) {
      b[i] = sWt[bcol * 64 + kk + half * 16 + i];
    }
    c = __builtin_amdgcn_wmma_f32_16x16x32_f16(false, a, false, b, (short)0, c,
                                               false, false);
  }
  // D layout: lane l, vgpr j -> row = rtile + 8*(l>>4) + j, col = ctile + (l&15)
  int trow = rtile + half * 8;
  int ocol = ctile + r;
#pragma unroll
  for (int j = 0; j < 8; ++j) sOut[(trow + j) * 64 + ocol] = c[j];
  __syncthreads();

  // coalesced writeback
  if (full) {
    float4* O4 = (float4*)(Out + (size_t)rowBase * 64);
    const float4* S4 = (const float4*)sOut;
    for (int i = tid; i < 32 * 16; i += 256) O4[i] = S4[i];
  } else {
    for (int i = tid; i < 32 * 64; i += 256) {
      int gr = rowBase + (i >> 6);
      if (gr < n) Out[(size_t)gr * 64 + (i & 63)] = sOut[i];
    }
  }
}

// ---------- 3) per-node attention dots: asrc[n]=ht.a_s, adst[n]=ht.a_d ----------
__global__ void dots_kernel(const float* __restrict__ ht, const float* __restrict__ av,
                            const float* __restrict__ dv, float* __restrict__ outS,
                            float* __restrict__ outD, int n) {
  int wave = threadIdx.x >> 5, lane = threadIdx.x & 31;
  int node = blockIdx.x * 8 + wave;
  if (node >= n) return;
  float x0 = ht[(size_t)node * 64 + lane], x1 = ht[(size_t)node * 64 + 32 + lane];
  float sa = waveSum(x0 * av[lane] + x1 * av[32 + lane]);
  float sd = waveSum(x0 * dv[lane] + x1 * dv[32 + lane]);
  if (lane == 0) { outS[node] = sa; outD[node] = sd; }
}

// ---------- 4) per-layer accumulator init ----------
__global__ void init_layer_kernel(float* acc, float* mx, float* den, int n) {
  int total = n * 64;
  for (int i = blockIdx.x * blockDim.x + threadIdx.x; i < total;
       i += gridDim.x * blockDim.x) {
    acc[i] = 0.0f;
    if (i < n) { mx[i] = -__builtin_inff(); den[i] = 0.0f; }
  }
}

// ---------- 5) edge softmax pass 1: segment max over dst ----------
__global__ void edge_max_kernel(const int* __restrict__ ei, const float* __restrict__ as,
                                const float* __restrict__ ad, float* __restrict__ mx,
                                int E, int n) {
  int Ep = E + n;
  for (int e = blockIdx.x * blockDim.x + threadIdx.x; e < Ep;
       e += gridDim.x * blockDim.x) {
    int s, d;
    if (e < E) { s = ei[e]; d = ei[E + e]; } else { s = d = e - E; }
    float l = as[s] + ad[d];
    l = (l > 0.0f) ? l : 0.2f * l;          // LeakyReLU(0.2)
    atomMaxF(&mx[d], l);
  }
}

// ---------- 6) edge softmax pass 2: exp + segment sum ----------
__global__ void edge_exp_kernel(const int* __restrict__ ei, const float* __restrict__ as,
                                const float* __restrict__ ad, const float* __restrict__ mx,
                                float* __restrict__ den, float* __restrict__ eexp,
                                int E, int n) {
  int Ep = E + n;
  for (int e = blockIdx.x * blockDim.x + threadIdx.x; e < Ep;
       e += gridDim.x * blockDim.x) {
    int s, d;
    if (e < E) { s = ei[e]; d = ei[E + e]; } else { s = d = e - E; }
    float l = as[s] + ad[d];
    l = (l > 0.0f) ? l : 0.2f * l;
    float ex = __expf(l - mx[d]);
    eexp[e] = ex;
    atomAddF(&den[d], ex);
  }
}

// ---------- 7) edge scatter: acc[dst] += alpha * ht[src] (wave per edge) ----------
__global__ void edge_scatter_kernel(const int* __restrict__ ei, const float* __restrict__ eexp,
                                    const float* __restrict__ den, const float* __restrict__ ht,
                                    float* __restrict__ acc, int E, int n) {
  int lane = threadIdx.x & 31;
  int wid = (int)((blockIdx.x * (unsigned)blockDim.x + threadIdx.x) >> 5);
  int Ep = E + n;
  if (wid >= Ep) return;
  int s, d;
  if (wid < E) { s = ei[wid]; d = ei[E + wid]; } else { s = d = wid - E; }
  float alpha = eexp[wid] / (den[d] + 1e-16f);
  atomAddF(&acc[(size_t)d * 64 + lane],      alpha * ht[(size_t)s * 64 + lane]);
  atomAddF(&acc[(size_t)d * 64 + 32 + lane], alpha * ht[(size_t)s * 64 + 32 + lane]);
}

// ---------- 8) bias + relu ----------
__global__ void bias_relu_kernel(const float* __restrict__ acc, const float* __restrict__ b,
                                 float* __restrict__ h, int n) {
  int total = n * 64;
  for (int i = blockIdx.x * blockDim.x + threadIdx.x; i < total;
       i += gridDim.x * blockDim.x)
    h[i] = fmaxf(acc[i] + b[i & 63], 0.0f);
}

// ---------- 9) gate = h @ Wg + bg ----------
__global__ void gate_kernel(const float* __restrict__ h, const float* __restrict__ Wg,
                            const float* __restrict__ bg, float* __restrict__ gate, int n) {
  int wave = threadIdx.x >> 5, lane = threadIdx.x & 31;
  int node = blockIdx.x * 8 + wave;
  if (node >= n) return;
  float s = waveSum(h[(size_t)node * 64 + lane] * Wg[lane] +
                    h[(size_t)node * 64 + 32 + lane] * Wg[32 + lane]);
  if (lane == 0) gate[node] = s + bg[0];
}

// ---------- 10) pooling init ----------
__global__ void init_pool_kernel(float* mxG, float* denG, float* pooled, int G) {
  int i = blockIdx.x * blockDim.x + threadIdx.x;
  if (i < G) { mxG[i] = -__builtin_inff(); denG[i] = 0.0f; }
  if (i < G * 64) pooled[i] = 0.0f;
}

__global__ void node_max_kernel(const float* __restrict__ gate, const int* __restrict__ batch,
                                float* __restrict__ mxG, int n) {
  for (int i = blockIdx.x * blockDim.x + threadIdx.x; i < n;
       i += gridDim.x * blockDim.x)
    atomMaxF(&mxG[batch[i]], gate[i]);
}

__global__ void node_exp_kernel(const float* __restrict__ gate, const int* __restrict__ batch,
                                const float* __restrict__ mxG, float* __restrict__ denG,
                                float* __restrict__ att, int n) {
  for (int i = blockIdx.x * blockDim.x + threadIdx.x; i < n;
       i += gridDim.x * blockDim.x) {
    int g = batch[i];
    float ex = __expf(gate[i] - mxG[g]);
    att[i] = ex;
    atomAddF(&denG[g], ex);
  }
}

// wave per node: pooled[g] += att * h[n]
__global__ void pool_scatter_kernel(const float* __restrict__ h, const int* __restrict__ batch,
                                    const float* __restrict__ att, const float* __restrict__ denG,
                                    float* __restrict__ pooled, int n) {
  int lane = threadIdx.x & 31;
  int node = (int)((blockIdx.x * (unsigned)blockDim.x + threadIdx.x) >> 5);
  if (node >= n) return;
  int g = batch[node];
  float a = att[node] / (denG[g] + 1e-16f);
  atomAddF(&pooled[g * 64 + lane],      a * h[(size_t)node * 64 + lane]);
  atomAddF(&pooled[g * 64 + 32 + lane], a * h[(size_t)node * 64 + 32 + lane]);
}

// ---------- 11) q head: relu(pooled@Wq1+bq1)@Wq2+bq2 ----------
__global__ void head_kernel(const float* __restrict__ pooled, const float* __restrict__ Wq1,
                            const float* __restrict__ bq1, const float* __restrict__ Wq2,
                            const float* __restrict__ bq2, float* __restrict__ out,
                            int G, int A) {
  __shared__ float sp[64 * 64];
  __shared__ float sq[64 * 64];
  int tid = threadIdx.x;
  for (int i = tid; i < G * 64; i += 256) sp[i] = pooled[i];
  __syncthreads();
  for (int i = tid; i < G * 64; i += 256) {
    int g = i >> 6, k = i & 63;
    float a = bq1[k];
#pragma unroll 8
    for (int j = 0; j < 64; ++j) a = fmaf(sp[g * 64 + j], Wq1[j * 64 + k], a);
    sq[i] = fmaxf(a, 0.0f);
  }
  __syncthreads();
  for (int i = tid; i < G * A; i += 256) {
    int g = i / A, q = i % A;
    float a = bq2[q];
#pragma unroll 8
    for (int j = 0; j < 64; ++j) a = fmaf(sq[g * 64 + j], Wq2[j * A + q], a);
    out[i] = a;
  }
}

// ---------- driver ----------
extern "C" void kernel_launch(void* const* d_in, const int* in_sizes, int n_in,
                              void* d_out, int out_size, void* d_ws, size_t ws_size,
                              hipStream_t stream) {
  const float* x    = (const float*)d_in[0];
  const int*   ei   = (const int*)d_in[1];
  const int*   batch= (const int*)d_in[2];
  const float* Wenc = (const float*)d_in[3];
  const float* benc = (const float*)d_in[4];
  const float* gln  = (const float*)d_in[5];
  const float* bln  = (const float*)d_in[6];
  const float* W1   = (const float*)d_in[7];
  const float* as1  = (const float*)d_in[8];
  const float* ad1  = (const float*)d_in[9];
  const float* b1   = (const float*)d_in[10];
  const float* W2   = (const float*)d_in[11];
  const float* as2  = (const float*)d_in[12];
  const float* ad2  = (const float*)d_in[13];
  const float* b2   = (const float*)d_in[14];
  const float* Wg   = (const float*)d_in[15];
  const float* bg   = (const float*)d_in[16];
  const float* Wq1  = (const float*)d_in[17];
  const float* bq1  = (const float*)d_in[18];
  const float* Wq2  = (const float*)d_in[19];
  const float* bq2  = (const float*)d_in[20];
  float* out = (float*)d_out;

  const int N  = in_sizes[0] / 5;
  const int E  = in_sizes[1] / 2;
  const int A  = in_sizes[20];
  const int G  = out_size / A;
  const int Ep = E + N;

  float* ws = (float*)d_ws;
  float* h      = ws;                              // N*64
  float* ht     = h    + (size_t)N * 64;           // N*64
  float* acc    = ht   + (size_t)N * 64;           // N*64
  float* asrc   = acc  + (size_t)N * 64;           // N
  float* adst   = asrc + N;                        // N
  float* mx     = adst + N;                        // N
  float* den    = mx   + N;                        // N
  float* eexp   = den  + N;                        // Ep
  float* mxG    = eexp + Ep;                       // G
  float* denG   = mxG  + G;                        // G
  float* pooled = denG + G;                        // G*64

  dim3 blk(256);
  const int nodeWaveBlocks   = (N + 7) / 8;
  const int edgeThreadBlocks = (Ep + 255) / 256;
  const int edgeWaveBlocks   = (Ep + 7) / 8;
  const int gsBlocks         = 1024;

  enc_kernel<<<nodeWaveBlocks, blk, 0, stream>>>(x, Wenc, benc, gln, bln, h, N);

  const float* Ws[2] = {W1, W2};
  const float* As[2] = {as1, as2};
  const float* Ad[2] = {ad1, ad2};
  const float* Bs[2] = {b1, b2};
  for (int layer = 0; layer < 2; ++layer) {
    init_layer_kernel<<<gsBlocks, blk, 0, stream>>>(acc, mx, den, N);
    gemm64_wmma<<<(N + 31) / 32, blk, 0, stream>>>(h, Ws[layer], ht, N);
    dots_kernel<<<nodeWaveBlocks, blk, 0, stream>>>(ht, As[layer], Ad[layer], asrc, adst, N);
    edge_max_kernel<<<edgeThreadBlocks, blk, 0, stream>>>(ei, asrc, adst, mx, E, N);
    edge_exp_kernel<<<edgeThreadBlocks, blk, 0, stream>>>(ei, asrc, adst, mx, den, eexp, E, N);
    edge_scatter_kernel<<<edgeWaveBlocks, blk, 0, stream>>>(ei, eexp, den, ht, acc, E, N);
    bias_relu_kernel<<<gsBlocks, blk, 0, stream>>>(acc, Bs[layer], h, N);
  }

  gate_kernel<<<nodeWaveBlocks, blk, 0, stream>>>(h, Wg, bg, asrc, N);
  init_pool_kernel<<<(G * 64 + 255) / 256, blk, 0, stream>>>(mxG, denG, pooled, G);
  node_max_kernel<<<(N + 255) / 256, blk, 0, stream>>>(asrc, batch, mxG, N);
  node_exp_kernel<<<(N + 255) / 256, blk, 0, stream>>>(asrc, batch, mxG, denG, adst, N);
  pool_scatter_kernel<<<nodeWaveBlocks, blk, 0, stream>>>(h, batch, adst, denG, pooled, N);
  head_kernel<<<1, blk, 0, stream>>>(pooled, Wq1, bq1, Wq2, bq2, out, G, A);
}
